// SNNRecurrent_67937792688237
// MI455X (gfx1250) — matrix operand order repaired and checked by reading, not compiled, not run on hardware
//
#include <hip/hip_runtime.h>

typedef __attribute__((ext_vector_type(16))) _Float16 v16h;
typedef __attribute__((ext_vector_type(8)))  _Float16 v8h;
typedef __attribute__((ext_vector_type(8)))  float    v8f;

#define TT  256
#define BB  64
#define NIN 1024
#define HH  2048

// ---------------------------------------------------------------------------
// WMMA fragment loaders (CDNA5 16x16x32 f16 layouts, CDNA5 ISA)
// A (MxK = 16x32): lanes 0-15  -> row M=lane,    K = k0+0..7  and k0+16..23
//                  lanes 16-31 -> row M=lane-16, K = k0+8..15 and k0+24..31
__device__ __forceinline__ v16h load_a_frag(const _Float16* __restrict__ A,
                                            int row0, int lda, int k0, int lane) {
    int r  = lane & 15;
    int kb = (lane & 16) ? 8 : 0;
    const _Float16* p = A + (size_t)(row0 + r) * lda + (k0 + kb);
    v8h lo = *(const v8h*)(p);        // 16B aligned: k0 mult 32, kb in {0,8}
    v8h hi = *(const v8h*)(p + 16);
    v16h a;
#pragma unroll
    for (int i = 0; i < 8; ++i) { a[i] = lo[i]; a[i + 8] = hi[i]; }
    return a;
}

// B (KxN = 32x16): lane holds col N=lane%16, K = k0+kb+0..7 and k0+kb+16..23 (kb = lane<16 ? 0 : 8),
// ascending K in the 16 halves. C[m][n] = sum_k A[m][k]*W[n][k], so
// B[k][n] = W[n][k] with W row-major over k -> two contiguous 16B runs per lane.
__device__ __forceinline__ v16h load_b_frag(const _Float16* __restrict__ W,
                                            int col0, int ldw, int k0, int lane) {
    int c  = lane & 15;
    int kb = (lane & 16) ? 8 : 0;
    const _Float16* p = W + (size_t)(col0 + c) * ldw + (k0 + kb);
    v8h lo = *(const v8h*)(p);
    v8h hi = *(const v8h*)(p + 16);
    v16h b;
#pragma unroll
    for (int i = 0; i < 8; ++i) { b[i] = lo[i]; b[i + 8] = hi[i]; }
    return b;
}

#define WMMA_F16(A, B, C) \
    __builtin_amdgcn_wmma_f32_16x16x32_f16(false, (A), false, (B), (short)0, (C), false, false)

// ---------------------------------------------------------------------------
__global__ void snn_cvt_f32_f16(const float* __restrict__ src,
                                _Float16* __restrict__ dst, int n) {
    int i = blockIdx.x * blockDim.x + threadIdx.x;
    if (i < n) dst[i] = (_Float16)src[i];
}

__global__ void snn_zero_f32(float* __restrict__ p, int n) {
    int i = blockIdx.x * blockDim.x + threadIdx.x;
    if (i < n) p[i] = 0.0f;
}

// ---------------------------------------------------------------------------
// Phase 1: pre[t,b,h] = x[t,b,:] @ w_in[h,:] as one big GEMM.
// rows R = T*B = 16384, cols H = 2048, K = 1024.
// 2x2 register tiling: each wave computes a 32x32 output block
// (2 A frags + 2 B frags -> 4 WMMAs per K chunk). 32768 wave-tiles.
__global__ void snn_input_gemm(const _Float16* __restrict__ X,
                               const _Float16* __restrict__ Wi,
                               float* __restrict__ pre) {
    int lane = threadIdx.x & 31;
    int wave = threadIdx.x >> 5;
    int tile = blockIdx.x * 8 + wave;       // 32768 tiles
    int mt = tile >> 6;                     // 512 row-groups of 32
    int nt = tile & 63;                     // 64 col-groups of 32
    int r0 = mt * 32, c0 = nt * 32;

    v8f acc00 = {}, acc01 = {}, acc10 = {}, acc11 = {};
#pragma unroll 2
    for (int k0 = 0; k0 < NIN; k0 += 32) {
        v16h a0 = load_a_frag(X, r0,      NIN, k0, lane);
        v16h a1 = load_a_frag(X, r0 + 16, NIN, k0, lane);
        v16h b0 = load_b_frag(Wi, c0,      NIN, k0, lane);
        v16h b1 = load_b_frag(Wi, c0 + 16, NIN, k0, lane);
        acc00 = WMMA_F16(a0, b0, acc00);
        acc01 = WMMA_F16(a0, b1, acc01);
        acc10 = WMMA_F16(a1, b0, acc10);
        acc11 = WMMA_F16(a1, b1, acc11);
    }
    int n  = lane & 15;
    int mb = (lane & 16) ? 8 : 0;
    auto st = [&](v8f acc, int rr, int cc) {
#pragma unroll
        for (int r = 0; r < 8; ++r)
            pre[(size_t)(r0 + rr + mb + r) * HH + (c0 + cc + n)] = acc[r];
    };
    st(acc00, 0, 0); st(acc01, 0, 16); st(acc10, 16, 0); st(acc11, 16, 16);
}

// ---------------------------------------------------------------------------
// Phase 2 (per step, fused): rec = z_prev @ Wrec^T + LIF update.
// B = 64 = four 16-row M-tiles: each wave owns ONE 16-col h-tile and all 4
// M-tiles -> each B (Wrec) fragment feeds 4 WMMAs, so Wrec is streamed from
// L2 exactly once per step (8 MB floor). 128 waves, 64 blocks of 2 waves.
// z double-buffered (zin read-only this step, zout written this step).
__global__ void snn_lif_step(const float* __restrict__ pre_t,
                             const _Float16* __restrict__ zin,
                             _Float16* __restrict__ zout,
                             const _Float16* __restrict__ Wr,
                             float* __restrict__ v, float* __restrict__ cur,
                             float* __restrict__ out_t,
                             float* __restrict__ fz, float* __restrict__ fv,
                             float* __restrict__ fi, int last) {
    int lane = threadIdx.x & 31;
    int wave = threadIdx.x >> 5;
    int ht = blockIdx.x * 2 + wave;         // 128 h-tile columns
    int h0 = ht * 16;

    v8f acc0 = {}, acc1 = {}, acc2 = {}, acc3 = {};
#pragma unroll 2
    for (int k0 = 0; k0 < HH; k0 += 32) {
        v16h b  = load_b_frag(Wr, h0, HH, k0, lane);
        v16h a0 = load_a_frag(zin,  0, HH, k0, lane);
        v16h a1 = load_a_frag(zin, 16, HH, k0, lane);
        v16h a2 = load_a_frag(zin, 32, HH, k0, lane);
        v16h a3 = load_a_frag(zin, 48, HH, k0, lane);
        acc0 = WMMA_F16(a0, b, acc0);
        acc1 = WMMA_F16(a1, b, acc1);
        acc2 = WMMA_F16(a2, b, acc2);
        acc3 = WMMA_F16(a3, b, acc3);
    }

    int n  = lane & 15;
    int mb = (lane & 16) ? 8 : 0;
    auto epi = [&](v8f acc, int b0) {
#pragma unroll
        for (int r = 0; r < 8; ++r) {
            int idx = (b0 + mb + r) * HH + (h0 + n);
            float vo = v[idx];
            float io = cur[idx];
            // v_decayed = v + dt*tau_mem_inv*((v_leak - v) + i) = 0.9 v + 0.1 i
            float vdec = 0.9f * vo + 0.1f * io;
            // i_decayed = i - dt*tau_syn_inv*i = 0.8 i
            float idec = 0.8f * io;
            float z  = (vdec > 1.0f) ? 1.0f : 0.0f;  // heaviside(vdec - v_th)
            float vn = (1.0f - z) * vdec;            // v_reset = 0
            float in = idec + pre_t[idx] + acc[r];   // + x@Win^T + z_prev@Wrec^T
            v[idx]    = vn;
            cur[idx]  = in;
            zout[idx] = (_Float16)z;
            out_t[idx] = z;
            if (last) { fz[idx] = z; fv[idx] = vn; fi[idx] = in; }
        }
    };
    epi(acc0, 0); epi(acc1, 16); epi(acc2, 32); epi(acc3, 48);
}

// ---------------------------------------------------------------------------
extern "C" void kernel_launch(void* const* d_in, const int* in_sizes, int n_in,
                              void* d_out, int out_size, void* d_ws, size_t ws_size,
                              hipStream_t stream) {
    (void)in_sizes; (void)n_in; (void)out_size; (void)ws_size;

    const float* x  = (const float*)d_in[0];   // [T,B,NIN] f32
    const float* wi = (const float*)d_in[1];   // [H,NIN]   f32
    const float* wr = (const float*)d_in[2];   // [H,H]     f32
    float* out = (float*)d_out;                // [T,B,H] spikes ++ (z,v,i) finals

    // workspace carve-out (256B aligned slots; f32 slots are 256B multiples,
    // so v,cur,z0,z1 are contiguous for a single zero-fill pass)
    char* ws = (char*)d_ws;
    size_t off = 0;
    auto carve = [&](size_t bytes) -> void* {
        void* p = ws + off;
        off += (bytes + 255) & ~(size_t)255;
        return p;
    };
    float*    pre = (float*)   carve((size_t)TT * BB * HH * 4);   // 128 MiB
    _Float16* xh  = (_Float16*)carve((size_t)TT * BB * NIN * 2);  //  32 MiB
    _Float16* wih = (_Float16*)carve((size_t)HH * NIN * 2);       //   4 MiB
    _Float16* wrh = (_Float16*)carve((size_t)HH * HH * 2);        //   8 MiB
    float*    v   = (float*)   carve((size_t)BB * HH * 4);
    float*    cur = (float*)   carve((size_t)BB * HH * 4);
    _Float16* z0  = (_Float16*)carve((size_t)BB * HH * 2);
    _Float16* z1  = (_Float16*)carve((size_t)BB * HH * 2);

    // f32 -> f16 conversions
    {
        int n = TT * BB * NIN;
        snn_cvt_f32_f16<<<(n + 255) / 256, 256, 0, stream>>>(x, xh, n);
        n = HH * NIN;
        snn_cvt_f32_f16<<<(n + 255) / 256, 256, 0, stream>>>(wi, wih, n);
        n = HH * HH;
        snn_cvt_f32_f16<<<(n + 255) / 256, 256, 0, stream>>>(wr, wrh, n);
    }

    // zero state (v, cur as f32; z0, z1 as f16 == one contiguous f32 region)
    {
        int n = 3 * BB * HH;   // (4+4+2+2)*BB*HH bytes / 4
        snn_zero_f32<<<(n + 255) / 256, 256, 0, stream>>>(v, n);
    }

    // Phase 1: bulk input GEMM, 32768 wave-tiles / 8 waves per block
    snn_input_gemm<<<(TT * BB / 32) * (HH / 32) / 8, 256, 0, stream>>>(xh, wih, pre);

    // Phase 2: 256 sequential fused recurrent-GEMM + LIF steps
    float* fz = out + (size_t)TT * BB * HH;
    float* fv = fz + (size_t)BB * HH;
    float* fi = fv + (size_t)BB * HH;
    for (int t = 0; t < TT; ++t) {
        _Float16* zin  = (t & 1) ? z1 : z0;
        _Float16* zout = (t & 1) ? z0 : z1;
        snn_lif_step<<<(HH / 16) / 2, 64, 0, stream>>>(
            pre + (size_t)t * BB * HH, zin, zout, wrh, v, cur,
            out + (size_t)t * BB * HH, fz, fv, fi, (t == TT - 1) ? 1 : 0);
    }
}
